// LocalBlock_31258771980818
// MI455X (gfx1250) — matrix operand order repaired and checked by reading, not compiled, not run on hardware
//
#include <hip/hip_runtime.h>
#include <cmath>

// ---------------------------------------------------------------------------
// Transformer block for MI455X (gfx1250, wave32, WMMA bf16 16x16x32).
// GEMMs + attention matmuls on v_wmma_f32_16x16x32_bf16, fp32 accumulate.
// GEMM tiles stream via GLOBAL_LOAD_ASYNC_TO_LDS_B128 (ASYNCcnt-tracked,
// double-buffered) when the builtin is available.
// ---------------------------------------------------------------------------

typedef __bf16 bf16_t;
typedef __attribute__((ext_vector_type(16))) __bf16 bf16x16;
typedef __attribute__((ext_vector_type(8))) float floatx8;
typedef __attribute__((ext_vector_type(4))) int i32x4;

#define D_MODEL 1024
#define NHEAD   16
#define HDIM    64
#define WINDOW  256
#define SEQ     2048
#define BATCH   2
#define TOKENS  (BATCH * SEQ)   // 4096

#define AS_GLOBAL __attribute__((address_space(1)))
#define AS_LDS    __attribute__((address_space(3)))

#if __has_builtin(__builtin_amdgcn_global_load_async_to_lds_b128)
#define HAVE_ASYNC_LDS 1
#else
#define HAVE_ASYNC_LDS 0
#endif

union Frag {
    uint4   u[2];
    bf16x16 v;
};

// 16B global -> LDS copy: async (no VGPR round trip) when available.
__device__ __forceinline__ void copy16_g2l(bf16_t* lds, const bf16_t* g) {
#if HAVE_ASYNC_LDS
    void* gp = (void*)g;   // drop const, then addrspace+type cast
    void* lp = (void*)lds;
    __builtin_amdgcn_global_load_async_to_lds_b128(
        (AS_GLOBAL i32x4*)gp, (AS_LDS i32x4*)lp, 0, 0);
#else
    *(uint4*)lds = *(const uint4*)g;
#endif
}

__device__ __forceinline__ void wait_async_le4() {
#if HAVE_ASYNC_LDS
#if __has_builtin(__builtin_amdgcn_s_wait_asynccnt)
    __builtin_amdgcn_s_wait_asynccnt(4);
#else
    __asm__ volatile("s_wait_asynccnt 0x4" ::: "memory");
#endif
#endif
}

__device__ __forceinline__ void wait_async_le0() {
#if HAVE_ASYNC_LDS
#if __has_builtin(__builtin_amdgcn_s_wait_asynccnt)
    __builtin_amdgcn_s_wait_asynccnt(0);
#else
    __asm__ volatile("s_wait_asynccnt 0x0" ::: "memory");
#endif
#endif
}

// ---------------------------------------------------------------------------
// fp32 [K,N] row-major -> bf16 [N,K] row-major (transposed weight)
// ---------------------------------------------------------------------------
__global__ __launch_bounds__(256)
void transpose_to_bf16(const float* __restrict__ W, bf16_t* __restrict__ WT,
                       int K, int N) {
    const size_t i = (size_t)blockIdx.x * 256 + threadIdx.x;
    if (i >= (size_t)K * N) return;
    const int n = (int)(i % N);
    const int k = (int)(i / N);
    WT[(size_t)n * K + k] = (bf16_t)W[i];
}

// ---------------------------------------------------------------------------
// LayerNorm (fp32 in) -> bf16 out.  One 256-thread block per row (d=1024).
// ---------------------------------------------------------------------------
__global__ __launch_bounds__(256)
void layernorm_to_bf16(const float* __restrict__ x, const float* __restrict__ sc,
                       const float* __restrict__ bi, bf16_t* __restrict__ y) {
    const int row = blockIdx.x;
    const float* xr = x + (size_t)row * D_MODEL;
    float v[4];
    float s = 0.f, ss = 0.f;
#pragma unroll
    for (int i = 0; i < 4; ++i) {
        v[i] = xr[threadIdx.x + i * 256];
        s += v[i];
        ss += v[i] * v[i];
    }
#pragma unroll
    for (int m = 1; m < 32; m <<= 1) {
        s  += __shfl_xor(s, m, 32);
        ss += __shfl_xor(ss, m, 32);
    }
    __shared__ float rs[8], rss[8];
    const int wave = threadIdx.x >> 5, lane = threadIdx.x & 31;
    if (lane == 0) { rs[wave] = s; rss[wave] = ss; }
    __syncthreads();
    if (wave == 0) {
        float a  = (lane < 8) ? rs[lane]  : 0.f;
        float b2 = (lane < 8) ? rss[lane] : 0.f;
#pragma unroll
        for (int m = 1; m < 8; m <<= 1) {
            a  += __shfl_xor(a, m, 32);
            b2 += __shfl_xor(b2, m, 32);
        }
        if (lane == 0) { rs[0] = a; rss[0] = b2; }
    }
    __syncthreads();
    const float mean = rs[0] * (1.f / D_MODEL);
    const float var  = rss[0] * (1.f / D_MODEL) - mean * mean;
    const float rstd = rsqrtf(var + 1e-5f);
#pragma unroll
    for (int i = 0; i < 4; ++i) {
        const int idx = threadIdx.x + i * 256;
        y[(size_t)row * D_MODEL + idx] = (bf16_t)((v[i] - mean) * rstd * sc[idx] + bi[idx]);
    }
}

// ---------------------------------------------------------------------------
// Tiled WMMA GEMM:  C[M,N] = A[M,K](bf16) x Bt[N,K](bf16)^T + bias, epilogue.
// 256 threads (8 waves), tile 128x128, K-step 32.  LDS rows padded to 80B.
// Double-buffered: tile k+1 streams via async-to-LDS while tile k is in the
// matrix units; per-wave s_wait_asynccnt + barrier for handoff.
//   EPI 0: QKV scatter -> q(*0.125), k ([b,h,s,64]) and vT ([b,h,64,s])
//   EPI 1/3: out = res + val (fp32)      EPI 2: out = gelu(val) (bf16)
// ---------------------------------------------------------------------------
template <int EPI>
__global__ __launch_bounds__(256)
void gemm_bf16(const bf16_t* __restrict__ A, const bf16_t* __restrict__ Bt,
               const float* __restrict__ bias, const float* __restrict__ res,
               float* __restrict__ outF, bf16_t* __restrict__ outB,
               bf16_t* __restrict__ outQ, bf16_t* __restrict__ outK,
               bf16_t* __restrict__ outVT, int M, int N, int K) {
    __shared__ __align__(16) bf16_t sA[2][128 * 40];
    __shared__ __align__(16) bf16_t sB[2][128 * 40];
    const int tid  = threadIdx.x;
    const int lane = tid & 31, l16 = lane & 15, half = lane >> 4;
    const int wave = tid >> 5;
    const int wm = wave >> 1;   // 0..3 : 32-row strip
    const int wn = wave & 1;    // 0..1 : 64-col strip
    const int rowBase = blockIdx.y * 128;
    const int colBase = blockIdx.x * 128;
    const int ldr = tid >> 2;          // 0..63
    const int ldc = (tid & 3) * 8;     // element offset 0,8,16,24

    const bf16_t* gA = A + (size_t)(rowBase + ldr) * K + ldc;
    const bf16_t* gB = Bt + (size_t)(colBase + ldr) * K + ldc;
    const size_t strideRow64A = (size_t)64 * K;

    floatx8 zero = {};
    floatx8 acc[2][4];
#pragma unroll
    for (int i = 0; i < 2; ++i)
#pragma unroll
        for (int j = 0; j < 4; ++j) acc[i][j] = zero;

    // Issue one 128x32 A-tile + 128x32 B-tile into buffer `buf` (4 async ops).
    auto issue_tile = [&](int kb, int buf) {
#pragma unroll
        for (int p = 0; p < 2; ++p) {
            const int row = p * 64 + ldr;
            copy16_g2l(&sA[buf][row * 40 + ldc], gA + (size_t)p * strideRow64A + kb);
            copy16_g2l(&sB[buf][row * 40 + ldc], gB + (size_t)p * strideRow64A + kb);
        }
    };

    issue_tile(0, 0);
    int buf = 0;
    for (int kb = 0; kb < K; kb += 32) {
        const bool more = (kb + 32) < K;
        if (more) issue_tile(kb + 32, buf ^ 1);   // prefetch into other buffer
        if (more) wait_async_le4(); else wait_async_le0();
        __syncthreads();

        Frag a[2], b[4];
#pragma unroll
        for (int ms = 0; ms < 2; ++ms) {   // A layout: K split 8/8 per half-wave
            const uint4* p = (const uint4*)(&sA[buf][(wm * 32 + ms * 16 + l16) * 40 + half * 8]);
            a[ms].u[0] = p[0];
            a[ms].u[1] = p[2];
        }
#pragma unroll
        for (int ns = 0; ns < 4; ++ns) {   // B layout: K split 8/8 per half-wave
            const uint4* p = (const uint4*)(&sB[buf][(wn * 64 + ns * 16 + l16) * 40 + half * 8]);
            b[ns].u[0] = p[0];
            b[ns].u[1] = p[2];
        }
#pragma unroll
        for (int ms = 0; ms < 2; ++ms)
#pragma unroll
            for (int ns = 0; ns < 4; ++ns)
                acc[ms][ns] = __builtin_amdgcn_wmma_f32_16x16x32_bf16(
                    false, a[ms].v, false, b[ns].v, (short)0, acc[ms][ns], false, false);
        __syncthreads();   // all waves done reading `buf` before it is refilled
        buf ^= 1;
    }

    // Epilogue.  C layout: lane -> N = l16, vgpr r -> M = r + 8*half.
#pragma unroll
    for (int ms = 0; ms < 2; ++ms) {
#pragma unroll
        for (int ns = 0; ns < 4; ++ns) {
            const int n  = colBase + wn * 64 + ns * 16 + l16;
            const float bv = bias[n];
#pragma unroll
            for (int r = 0; r < 8; ++r) {
                const int m = rowBase + wm * 32 + ms * 16 + r + 8 * half;
                const float val = acc[ms][ns][r] + bv;
                if (EPI == 0) {
                    const int bidx = m >> 11, sPos = m & 2047;
                    if (n < 1024) {
                        const int hh = n >> 6, dd = n & 63;
                        outQ[((size_t)(bidx * NHEAD + hh) * SEQ + sPos) * HDIM + dd] =
                            (bf16_t)(val * 0.125f);  // 1/sqrt(HDIM) folded into Q
                    } else if (n < 2048) {
                        const int n2 = n - 1024, hh = n2 >> 6, dd = n2 & 63;
                        outK[((size_t)(bidx * NHEAD + hh) * SEQ + sPos) * HDIM + dd] = (bf16_t)val;
                    } else {
                        const int n2 = n - 2048, hh = n2 >> 6, dd = n2 & 63;
                        outVT[((size_t)(bidx * NHEAD + hh) * HDIM + dd) * SEQ + sPos] = (bf16_t)val;
                    }
                } else if (EPI == 1) {
                    outF[(size_t)m * N + n] = res[(size_t)m * N + n] + val;
                } else if (EPI == 2) {
                    const float g = 0.5f * val * (1.f + erff(val * 0.70710678f));
                    outB[(size_t)m * N + n] = (bf16_t)g;
                } else {
                    outF[(size_t)m * N + n] = res[(size_t)m * N + n] + val;
                }
            }
        }
    }
}

// ---------------------------------------------------------------------------
// Sliding-window flash attention.  128 threads = 4 independent waves, each
// wave owns a 16-query tile; 32 keys/iter: S = Q K^T (4 WMMA), online softmax
// (shfl_xor row reductions in the C layout), P restaged via per-wave LDS into
// an A fragment, O += P V (4 WMMA against pre-transposed V^T).  Interior
// chunks (wave-uniformly inside the window) skip all masking VALU.
// ---------------------------------------------------------------------------
__global__ __launch_bounds__(128)
void attn_kernel(const bf16_t* __restrict__ Qg, const bf16_t* __restrict__ Kg,
                 const bf16_t* __restrict__ VTg, bf16_t* __restrict__ out) {
    __shared__ __align__(16) bf16_t sP[4][16 * 40];
    const int tid = threadIdx.x;
    const int wave = tid >> 5, lane = tid & 31;
    const int l16 = lane & 15, half = lane >> 4;
    const int h = blockIdx.y, b = blockIdx.z;
    const int q0 = (blockIdx.x * 4 + wave) * 16;
    const size_t headoff = (size_t)(b * NHEAD + h) * SEQ * HDIM;
    const bf16_t* q  = Qg + headoff;
    const bf16_t* kp = Kg + headoff;
    const bf16_t* vt = VTg + headoff;   // [HDIM, SEQ] within head
    bf16_t* sPw = sP[wave];

    Frag aQ[2];
#pragma unroll
    for (int kk = 0; kk < 2; ++kk) {    // A-fragment layout
        const uint4* p = (const uint4*)(q + (size_t)(q0 + l16) * HDIM + kk * 32 + half * 8);
        aQ[kk].u[0] = p[0];
        aQ[kk].u[1] = p[2];
    }

    float m_i[8], l_i[8];
    floatx8 zero = {};
    floatx8 acc[4];
#pragma unroll
    for (int r = 0; r < 8; ++r) { m_i[r] = -30000.f; l_i[r] = 0.f; }
#pragma unroll
    for (int j = 0; j < 4; ++j) acc[j] = zero;

    int kstart = q0 - WINDOW;
    if (kstart < 0) kstart = 0;
    kstart &= ~31;
    const int kend = q0 + 15;

    for (int k0 = kstart; k0 <= kend; k0 += 32) {
        floatx8 s0 = zero, s1 = zero;
#pragma unroll
        for (int kk = 0; kk < 2; ++kk) {
            Frag bK0, bK1;   // B-fragment: lane = key column, d split 8/8 per half
            const uint4* p0p = (const uint4*)(kp + (size_t)(k0 + l16) * HDIM + kk * 32 + half * 8);
            bK0.u[0] = p0p[0];
            bK0.u[1] = p0p[2];
            const uint4* p1p = (const uint4*)(kp + (size_t)(k0 + 16 + l16) * HDIM + kk * 32 + half * 8);
            bK1.u[0] = p1p[0];
            bK1.u[1] = p1p[2];
            s0 = __builtin_amdgcn_wmma_f32_16x16x32_bf16(false, aQ[kk].v, false, bK0.v,
                                                         (short)0, s0, false, false);
            s1 = __builtin_amdgcn_wmma_f32_16x16x32_bf16(false, aQ[kk].v, false, bK1.v,
                                                         (short)0, s1, false, false);
        }
        // Wave-uniform: is the whole 16x32 tile inside the allowed window?
        const bool fullTile = (k0 + 31 <= q0) && ((q0 + 15 - k0) <= WINDOW);
        // Online softmax over this 32-key chunk.
#pragma unroll
        for (int r = 0; r < 8; ++r) {
            float s0v = s0[r], s1v = s1[r];
            if (!fullTile) {
                const int m  = q0 + r + 8 * half;
                const int n0 = k0 + l16, n1 = n0 + 16;
                const int d0 = m - n0, d1 = m - n1;
                s0v = (d0 >= 0 && d0 <= WINDOW) ? s0v : -30000.f;
                s1v = (d1 >= 0 && d1 <= WINDOW) ? s1v : -30000.f;
            }
            float mx = fmaxf(s0v, s1v);
#pragma unroll
            for (int msk = 1; msk < 16; msk <<= 1) mx = fmaxf(mx, __shfl_xor(mx, msk, 32));
            const float mnew  = fmaxf(m_i[r], mx);
            const float alpha = __expf(m_i[r] - mnew);
            const float p0 = __expf(s0v - mnew);
            const float p1 = __expf(s1v - mnew);
            float ps = p0 + p1;
#pragma unroll
            for (int msk = 1; msk < 16; msk <<= 1) ps += __shfl_xor(ps, msk, 32);
            l_i[r] = l_i[r] * alpha + ps;
            m_i[r] = mnew;
#pragma unroll
            for (int j = 0; j < 4; ++j) acc[j][r] *= alpha;
            const int row = r + 8 * half;
            sPw[row * 40 + l16]      = (bf16_t)p0;
            sPw[row * 40 + 16 + l16] = (bf16_t)p1;
        }
        __asm__ volatile("s_wait_dscnt 0x0" ::: "memory");
        Frag aP;
        {
            const uint4* p = (const uint4*)(&sPw[l16 * 40 + half * 8]);
            aP.u[0] = p[0];
            aP.u[1] = p[2];
        }
#pragma unroll
        for (int j = 0; j < 4; ++j) {
            Frag bV;   // V^T: row = d, keys split 8/8 per half
            const uint4* p = (const uint4*)(vt + (size_t)(j * 16 + l16) * SEQ + k0 + half * 8);
            bV.u[0] = p[0];
            bV.u[1] = p[2];
            acc[j] = __builtin_amdgcn_wmma_f32_16x16x32_bf16(false, aP.v, false, bV.v,
                                                             (short)0, acc[j], false, false);
        }
    }

#pragma unroll
    for (int r = 0; r < 8; ++r) {
        const float inv = 1.f / l_i[r];
        const int m = q0 + r + 8 * half;
#pragma unroll
        for (int j = 0; j < 4; ++j) {
            const int col = h * HDIM + j * 16 + l16;
            out[(size_t)(b * SEQ + m) * D_MODEL + col] = (bf16_t)(acc[j][r] * inv);
        }
    }
}

// ---------------------------------------------------------------------------
extern "C" void kernel_launch(void* const* d_in, const int* in_sizes, int n_in,
                              void* d_out, int out_size, void* d_ws, size_t ws_size,
                              hipStream_t stream) {
    const float* hidden = (const float*)d_in[0];
    const float* n1s    = (const float*)d_in[1];
    const float* n1b    = (const float*)d_in[2];
    const float* w_qkv  = (const float*)d_in[3];
    const float* b_qkv  = (const float*)d_in[4];
    const float* w_out  = (const float*)d_in[5];
    const float* b_out  = (const float*)d_in[6];
    const float* n2s    = (const float*)d_in[7];
    const float* n2b    = (const float*)d_in[8];
    const float* w_mlp1 = (const float*)d_in[9];
    const float* b_mlp1 = (const float*)d_in[10];
    const float* w_mlp2 = (const float*)d_in[11];
    const float* b_mlp2 = (const float*)d_in[12];
    float* out = (float*)d_out;

    char* ws = (char*)d_ws;
    size_t off = 0;
    auto alloc = [&](size_t bytes) -> void* {
        void* p = ws + off;
        off = (off + bytes + 255) & ~(size_t)255;
        return p;
    };
    bf16_t* wqkvT  = (bf16_t*)alloc((size_t)3072 * 1024 * 2);
    bf16_t* woutT  = (bf16_t*)alloc((size_t)1024 * 1024 * 2);
    bf16_t* wm1T   = (bf16_t*)alloc((size_t)4096 * 1024 * 2);
    bf16_t* wm2T   = (bf16_t*)alloc((size_t)1024 * 4096 * 2);
    bf16_t* norm1  = (bf16_t*)alloc((size_t)TOKENS * D_MODEL * 2);
    bf16_t* qbuf   = (bf16_t*)alloc((size_t)TOKENS * D_MODEL * 2);
    bf16_t* kbuf   = (bf16_t*)alloc((size_t)TOKENS * D_MODEL * 2);
    bf16_t* vtbuf  = (bf16_t*)alloc((size_t)TOKENS * D_MODEL * 2);
    bf16_t* attbuf = (bf16_t*)alloc((size_t)TOKENS * D_MODEL * 2);
    float*  hbuf   = (float*)alloc((size_t)TOKENS * D_MODEL * 4);
    bf16_t* norm2  = (bf16_t*)alloc((size_t)TOKENS * D_MODEL * 2);
    bf16_t* actbuf = (bf16_t*)alloc((size_t)TOKENS * 4 * D_MODEL * 2);
    (void)ws_size; (void)in_sizes; (void)n_in; (void)out_size;

    // 1) weight prep (fp32 -> bf16, transposed to [N,K])
    transpose_to_bf16<<<(1024 * 3072 + 255) / 256, 256, 0, stream>>>(w_qkv, wqkvT, 1024, 3072);
    transpose_to_bf16<<<(1024 * 1024 + 255) / 256, 256, 0, stream>>>(w_out, woutT, 1024, 1024);
    transpose_to_bf16<<<(1024 * 4096 + 255) / 256, 256, 0, stream>>>(w_mlp1, wm1T, 1024, 4096);
    transpose_to_bf16<<<(4096 * 1024 + 255) / 256, 256, 0, stream>>>(w_mlp2, wm2T, 4096, 1024);

    // 2) LN1
    layernorm_to_bf16<<<TOKENS, 256, 0, stream>>>(hidden, n1s, n1b, norm1);

    // 3) QKV projection, scatter into head layouts (V transposed)
    gemm_bf16<0><<<dim3(3072 / 128, TOKENS / 128), 256, 0, stream>>>(
        norm1, wqkvT, b_qkv, nullptr, nullptr, nullptr, qbuf, kbuf, vtbuf,
        TOKENS, 3072, 1024);

    // 4) sliding-window attention
    attn_kernel<<<dim3(SEQ / 64, NHEAD, BATCH), 128, 0, stream>>>(qbuf, kbuf, vtbuf, attbuf);

    // 5) out projection + residual -> h (fp32)
    gemm_bf16<1><<<dim3(1024 / 128, TOKENS / 128), 256, 0, stream>>>(
        attbuf, woutT, b_out, hidden, hbuf, nullptr, nullptr, nullptr, nullptr,
        TOKENS, 1024, 1024);

    // 6) LN2
    layernorm_to_bf16<<<TOKENS, 256, 0, stream>>>(hbuf, n2s, n2b, norm2);

    // 7) MLP up + exact GELU -> bf16
    gemm_bf16<2><<<dim3(4096 / 128, TOKENS / 128), 256, 0, stream>>>(
        norm2, wm1T, b_mlp1, nullptr, nullptr, actbuf, nullptr, nullptr, nullptr,
        TOKENS, 4096, 1024);

    // 8) MLP down + residual -> d_out (fp32)
    gemm_bf16<3><<<dim3(1024 / 128, TOKENS / 128), 256, 0, stream>>>(
        actbuf, wm2T, b_mlp2, hbuf, out, nullptr, nullptr, nullptr, nullptr,
        TOKENS, 1024, 4096);
}
